// Compressor_72060961292371
// MI455X (gfx1250) — matrix operand order repaired and checked by reading, not compiled, not run on hardware
//
#include <hip/hip_runtime.h>
#include <hip/hip_bf16.h>

// ---------------------------------------------------------------------------
// MI455X (gfx1250) fused "Compressor" kernel.
//   GEMM path: bf16 WMMA (v_wmma_f32_16x16x32_bf16); x (235MB f32) is read
//   from HBM exactly once, converted on the fly through double-buffered LDS.
//   sched_barrier #1 pins all 16 B-fragment loads + next-step x loads ahead
//   of the WMMA block (staggered s_wait_loadcnt, no per-WMMA drain).
//   sched_barrier #2 pins the x convert/ds_store AFTER the WMMA block so the
//   HBM-latency x loads retire under all 16 WMMAs, not after 2.
//   Epilogue: overlap-softmax + RMSnorm + RoPE + (collapsed) FWHT, one wave32
//   per output block using cross-lane __shfl_xor (wave32-correct).
// ---------------------------------------------------------------------------

#define DIM   7168
#define SEQ   8192
#define NB    2048
#define OUTC  256      // COFF*HD
#define HD    128
#define RD    64
#define EPSV  1e-6f
#define NEGV  (-1000000000.0f)

typedef __attribute__((ext_vector_type(16))) __bf16 v16bf;
typedef __attribute__((ext_vector_type(8)))  __bf16 v8bf;
typedef __attribute__((ext_vector_type(8)))  float  v8f;

__device__ __forceinline__ unsigned short f2bf_bits(float f) {
  unsigned u = __builtin_bit_cast(unsigned, f);
  u += 0x7FFFu + ((u >> 16) & 1u);            // round-to-nearest-even
  return (unsigned short)(u >> 16);
}
__device__ __forceinline__ __bf16 f2bf(float f) {
  unsigned short h = f2bf_bits(f);
  return __builtin_bit_cast(__bf16, h);
}

// ---------------------------------------------------------------------------
// Kernel 1: convert a 256x7168 f32 weight matrix to bf16 (row-major kept;
// K-contiguous rows are exactly the per-lane B-fragment layout).
// ---------------------------------------------------------------------------
__global__ void __launch_bounds__(256)
cvt_bf16_kernel(const float* __restrict__ src, unsigned short* __restrict__ dst, int n4) {
  int i = blockIdx.x * blockDim.x + threadIdx.x;
  if (i >= n4) return;
  const float4 f = reinterpret_cast<const float4*>(src)[i];
  ushort4 o;
  o.x = f2bf_bits(f.x); o.y = f2bf_bits(f.y);
  o.z = f2bf_bits(f.z); o.w = f2bf_bits(f.w);
  reinterpret_cast<ushort4*>(dst)[i] = o;
}

// ---------------------------------------------------------------------------
// Kernel 2: kv = x*Wkv^T and score = x*Wgate^T via WMMA bf16.
//   grid = SEQ/64 workgroups, 256 threads (8 wave32).
//   wave w: m-tile = w&3 (16 rows each), sel = w>>2 (0 -> kv, 1 -> score).
//   Software-pipelined: LDS x-tile is double-buffered, one barrier per K-step.
// ---------------------------------------------------------------------------
#define LDSTR 40   // staged row stride in bf16 elems (32 data + 8 pad, 16B-aligned chunks)

__global__ void __launch_bounds__(256)
gemm_kv_score_kernel(const float* __restrict__ x,
                     const unsigned short* __restrict__ wkv_bf,
                     const unsigned short* __restrict__ wg_bf,
                     float* __restrict__ kv_out,
                     float* __restrict__ sc_out) {
  __shared__ __align__(16) __bf16 ldsA[2][64 * LDSTR];

  const int tid  = threadIdx.x;
  const int wave = tid >> 5;
  const int lane = tid & 31;
  const int mt   = wave & 3;       // which 16-row tile of the 64-row slab
  const int sel  = wave >> 2;      // 0 = kv, 1 = score
  const int lo   = lane & 15;
  const int hi   = lane >> 4;

  const int row0 = blockIdx.x * 64;

  // cooperative staging map: each thread converts 8 f32 of one row per step
  const int srow = tid >> 2;
  const int scol = (tid & 3) * 8;
  const float* srcp = x + (size_t)(row0 + srow) * DIM + scol;
  __bf16* const stage0 = &ldsA[0][srow * LDSTR + scol];
  __bf16* const stage1 = &ldsA[1][srow * LDSTR + scol];

  const __bf16* W =
      reinterpret_cast<const __bf16*>(sel ? wg_bf : wkv_bf);
  const int arow = mt * 16 + lo;   // A-fragment row inside the LDS slab

  v8f acc[16];
  #pragma unroll
  for (int t = 0; t < 16; ++t) acc[t] = v8f{0.f,0.f,0.f,0.f,0.f,0.f,0.f,0.f};

  // ---- prologue: stage K-step 0 into buffer 0 ------------------------------
  {
    const float4 f0 = *reinterpret_cast<const float4*>(srcp);
    const float4 f1 = *reinterpret_cast<const float4*>(srcp + 4);
    v8bf v;
    v[0] = f2bf(f0.x); v[1] = f2bf(f0.y); v[2] = f2bf(f0.z); v[3] = f2bf(f0.w);
    v[4] = f2bf(f1.x); v[5] = f2bf(f1.y); v[6] = f2bf(f1.z); v[7] = f2bf(f1.w);
    *reinterpret_cast<v8bf*>(stage0) = v;
  }
  __syncthreads();

  // ---- main pipelined loop -------------------------------------------------
  for (int k0 = 0, it = 0; k0 < DIM; k0 += 32, ++it) {
    const int cur = it & 1;
    const bool have_next = (k0 + 32) < DIM;

    // (1) A fragment from current buffer (contiguous k order, same as B):
    //     lanes 0-15 (hi=0): K = 0..15 of row `arow`
    //     lanes 16-31 (hi=1): K = 16..31
    const __bf16* abase = &ldsA[cur][arow * LDSTR];
    const v8bf a_lo = *reinterpret_cast<const v8bf*>(abase + 16 * hi);
    const v8bf a_hi = *reinterpret_cast<const v8bf*>(abase + 16 * hi + 8);
    const v16bf a = __builtin_shufflevector(a_lo, a_hi,
        0,1,2,3,4,5,6,7,8,9,10,11,12,13,14,15);

    // (2) preload all 16 B fragments for this K-step into registers.
    //     B layout: lane n holds 16 *contiguous* K values of column n
    //     (lanes 0-15: K = k0..k0+15 ; lanes 16-31: K = k0+16..k0+31)
    const __bf16* wbase = W + (size_t)lo * DIM + k0 + 16 * hi;
    v16bf b[16];
    #pragma unroll
    for (int t = 0; t < 16; ++t)
      b[t] = *reinterpret_cast<const v16bf*>(wbase + (size_t)t * 16 * DIM);

    // (3) issue next-step x loads last: youngest outstanding loads, so no
    //     WMMA wait threshold forces their (HBM-latency) completion.
    float4 nf0, nf1;
    if (have_next) {
      nf0 = *reinterpret_cast<const float4*>(srcp + k0 + 32);
      nf1 = *reinterpret_cast<const float4*>(srcp + k0 + 36);
    }

    // ---- fence #1: loads cannot be sunk into the WMMA block, so all
    //      fragments stay live and the waitcnt pass staggers the waits.
    __builtin_amdgcn_sched_barrier(0);

    // (4) 16 WMMAs
    #pragma unroll
    for (int t = 0; t < 16; ++t) {
      acc[t] = __builtin_amdgcn_wmma_f32_16x16x32_bf16(
          /*neg_a=*/false, a, /*neg_b=*/false, b[t],
          /*c_mod=*/(short)0, acc[t], /*reuse_a=*/false, /*reuse_b=*/false);
    }

    // ---- fence #2: the x convert (and its s_wait_loadcnt 0 drain) must not
    //      be hoisted into the WMMA block; x HBM latency hides under 16 WMMAs.
    __builtin_amdgcn_sched_barrier(0);

    // (5) convert and stage next step into the other buffer
    if (have_next) {
      v8bf v;
      v[0] = f2bf(nf0.x); v[1] = f2bf(nf0.y); v[2] = f2bf(nf0.z); v[3] = f2bf(nf0.w);
      v[4] = f2bf(nf1.x); v[5] = f2bf(nf1.y); v[6] = f2bf(nf1.z); v[7] = f2bf(nf1.w);
      *reinterpret_cast<v8bf*>(cur ? stage0 : stage1) = v;
    }

    // (6) single barrier per K-step: staged writes visible, previous reads done
    __syncthreads();
  }

  // C/D layout (7.12.2): lanes 0-15 -> N=lo, M=r; lanes 16-31 -> N=lo, M=r+8
  float* dst = sel ? sc_out : kv_out;
  const int rbase = row0 + mt * 16 + 8 * hi;
  #pragma unroll
  for (int r = 0; r < 8; ++r) {
    float* drow = dst + (size_t)(rbase + r) * OUTC + lo;
    #pragma unroll
    for (int t = 0; t < 16; ++t) drow[t * 16] = acc[t][r];
  }
}

// ---------------------------------------------------------------------------
// Kernel 3: overlap + softmax(8) + RMSnorm(128) + RoPE(64) + FWHT.
//   One wave32 per block n; lane l owns h = 4l..4l+3.
//   Reference FWHT splits into *halves* every step => each step is the same
//   top-level butterfly B (B^2 = 2I); 7 steps on d=128 -> 8*B; with the
//   128^-1/2 scale the transform is exactly out = B*x / sqrt(2).
// ---------------------------------------------------------------------------
__global__ void __launch_bounds__(256)
epilogue_kernel(const float* __restrict__ kv,
                const float* __restrict__ sc,
                const float* __restrict__ ape,
                const float* __restrict__ norm_w,
                const float* __restrict__ cosb,
                const float* __restrict__ sinb,
                float* __restrict__ out) {
  const int wave = threadIdx.x >> 5;
  const int lane = threadIdx.x & 31;
  const int n = blockIdx.x * 8 + wave;

  float kvb[8][4], scb[8][4];

  // rows 0..3: first half (cols 0..127) of the *previous* block, or fill
  #pragma unroll
  for (int r = 0; r < 4; ++r) {
    if (n > 0) {
      const int s = 4 * (n - 1) + r;
      const float4 k4 = *reinterpret_cast<const float4*>(kv + (size_t)s * OUTC + 4 * lane);
      const float4 s4 = *reinterpret_cast<const float4*>(sc + (size_t)s * OUTC + 4 * lane);
      const float4 a4 = *reinterpret_cast<const float4*>(ape + r * OUTC + 4 * lane);
      kvb[r][0] = k4.x + a4.x; kvb[r][1] = k4.y + a4.y;
      kvb[r][2] = k4.z + a4.z; kvb[r][3] = k4.w + a4.w;
      scb[r][0] = s4.x; scb[r][1] = s4.y; scb[r][2] = s4.z; scb[r][3] = s4.w;
    } else {
      #pragma unroll
      for (int c = 0; c < 4; ++c) { kvb[r][c] = 0.f; scb[r][c] = NEGV; }
    }
  }
  // rows 4..7: second half (cols 128..255) of the current block
  #pragma unroll
  for (int j = 0; j < 4; ++j) {
    const int s = 4 * n + j;
    const int col = HD + 4 * lane;
    const float4 k4 = *reinterpret_cast<const float4*>(kv + (size_t)s * OUTC + col);
    const float4 s4 = *reinterpret_cast<const float4*>(sc + (size_t)s * OUTC + col);
    const float4 a4 = *reinterpret_cast<const float4*>(ape + j * OUTC + col);
    kvb[4+j][0] = k4.x + a4.x; kvb[4+j][1] = k4.y + a4.y;
    kvb[4+j][2] = k4.z + a4.z; kvb[4+j][3] = k4.w + a4.w;
    scb[4+j][0] = s4.x; scb[4+j][1] = s4.y; scb[4+j][2] = s4.z; scb[4+j][3] = s4.w;
  }

  // softmax over the 8 overlapped rows, weighted sum -> comp
  float comp[4];
  float sumsq = 0.f;
  #pragma unroll
  for (int c = 0; c < 4; ++c) {
    float m = scb[0][c];
    #pragma unroll
    for (int r = 1; r < 8; ++r) m = fmaxf(m, scb[r][c]);
    float den = 0.f, num = 0.f;
    #pragma unroll
    for (int r = 0; r < 8; ++r) {
      const float e = __expf(scb[r][c] - m);
      den += e;
      num += e * kvb[r][c];
    }
    comp[c] = num / den;
    sumsq += comp[c] * comp[c];
  }

  // RMS norm across 128 (wave32 all-reduce)
  #pragma unroll
  for (int off = 16; off >= 1; off >>= 1) sumsq += __shfl_xor(sumsq, off, 32);
  const float scale = rsqrtf(sumsq * (1.0f / HD) + EPSV);
  const float4 nw = *reinterpret_cast<const float4*>(norm_w + 4 * lane);
  comp[0] *= scale * nw.x; comp[1] *= scale * nw.y;
  comp[2] *= scale * nw.z; comp[3] *= scale * nw.w;

  // RoPE on first RD=64 dims: lane l<16 owns pairs (2l, 2l+1) in-lane
  if (lane < 16) {
    const float c0 = cosb[n * 32 + 2 * lane],     s0 = sinb[n * 32 + 2 * lane];
    const float c1 = cosb[n * 32 + 2 * lane + 1], s1 = sinb[n * 32 + 2 * lane + 1];
    const float a0 = comp[0], a1 = comp[1];
    comp[0] = a0 * c0 - a1 * s0; comp[1] = a0 * s0 + a1 * c0;
    const float a2 = comp[2], a3 = comp[3];
    comp[2] = a2 * c1 - a3 * s1; comp[3] = a2 * s1 + a3 * c1;
  }

  // collapsed FWHT: out[h] = (x[h] + x[h+64])/sqrt2 ; out[h+64] = (x[h]-x[h+64])/sqrt2
  #pragma unroll
  for (int c = 0; c < 4; ++c) {
    const float other = __shfl_xor(comp[c], 16, 32);
    comp[c] = ((lane < 16) ? (comp[c] + other) : (other - comp[c]))
              * 0.70710678118654752440f;
  }

  float4 o; o.x = comp[0]; o.y = comp[1]; o.z = comp[2]; o.w = comp[3];
  *reinterpret_cast<float4*>(out + (size_t)n * HD + 4 * lane) = o;
}

// ---------------------------------------------------------------------------
// Launch
// ---------------------------------------------------------------------------
extern "C" void kernel_launch(void* const* d_in, const int* in_sizes, int n_in,
                              void* d_out, int out_size, void* d_ws, size_t ws_size,
                              hipStream_t stream) {
  (void)in_sizes; (void)n_in; (void)out_size; (void)ws_size;

  const float* x       = (const float*)d_in[0];
  const float* ape     = (const float*)d_in[1];
  const float* wkv_w   = (const float*)d_in[2];
  const float* wgate_w = (const float*)d_in[3];
  const float* norm_w  = (const float*)d_in[4];
  const float* cosb    = (const float*)d_in[5];
  const float* sinb    = (const float*)d_in[6];
  float* out = (float*)d_out;

  // workspace layout (needs ~24 MB)
  char* ws = (char*)d_ws;
  unsigned short* wkv_bf = (unsigned short*)(ws);                 // 3.67 MB
  unsigned short* wg_bf  = (unsigned short*)(ws + (4u  << 20));   // 3.67 MB
  float*          kv_ws  = (float*)(ws + (8u  << 20));            // 8.39 MB
  float*          sc_ws  = (float*)(ws + (16u << 20));            // 8.39 MB

  {
    const int n4  = (OUTC * DIM) / 4;
    const int blk = 256, grd = (n4 + blk - 1) / blk;
    cvt_bf16_kernel<<<grd, blk, 0, stream>>>(wkv_w,   wkv_bf, n4);
    cvt_bf16_kernel<<<grd, blk, 0, stream>>>(wgate_w, wg_bf,  n4);
  }
  gemm_kv_score_kernel<<<SEQ / 64, 256, 0, stream>>>(x, wkv_bf, wg_bf, kv_ws, sc_ws);
  epilogue_kernel<<<NB / 8, 256, 0, stream>>>(kv_ws, sc_ws, ape, norm_w, cosb, sinb, out);
}
